// _MultiHeadAttention_49374944035650
// MI455X (gfx1250) — matrix operand order repaired and checked by reading, not compiled, not run on hardware
//
#include <hip/hip_runtime.h>
#include <hip/hip_bf16.h>

// ---------------------------------------------------------------------------
// GQA attention for MI455X (gfx1250, wave32, WMMA).
// - All GEMM-like math on v_wmma_f32_16x16x32_bf16 (fp32 accumulate): the
//   problem is compute-bound (~530 FLOP/byte vs 23.3 TB/s HBM) and the bf16
//   WMMA path has 8x the K-depth/instruction of the f32 path.
// - AV stage stages K/V tiles into LDS with GLOBAL_LOAD_ASYNC_TO_LDS_B128,
//   double-buffered (s_wait_asynccnt 16) to overlap copy with WMMA.
// ---------------------------------------------------------------------------

typedef __bf16 v16bf __attribute__((ext_vector_type(16)));
typedef float  v8f   __attribute__((ext_vector_type(8)));

#define WMMA_BF16(a, b, c) \
  __builtin_amdgcn_wmma_f32_16x16x32_bf16(false, (a), false, (b), (short)0, (c), false, false)

static constexpr int B_  = 2;
static constexpr int N_  = 2048;
static constexpr int E_  = 1024;
static constexpr int H_  = 16;
static constexpr int G_  = 4;
static constexpr int D_  = 64;
static constexpr int HPG_ = H_ / G_;          // 4
static constexpr int GD_  = G_ * D_;          // 256
static constexpr int BN_  = B_ * N_;          // 4096
static constexpr float SCALE_ = 0.125f;       // 64^-0.5

__device__ __forceinline__ __bf16 to_bf16(float x)  { return (__bf16)x; }
__device__ __forceinline__ __bf16 to_bf16(__bf16 x) { return x; }

// ---------------------------------------------------------------------------
// Generic C = A * W^T + bias GEMM. A: [M,K] (fp32 or bf16), W: [N,K] fp32
// row-major, C: [M,N] (fp32 or bf16). One wave -> one 16x16 C tile.
// A-fragment (16-bit, 16x32): lane half=l/16, row m=l%16:
//   k = half*16 + j
// B-fragment (32x16): col n = l%16; lanes0-15 hold k=0..15, lanes16-31 k=16..31
// C layout: element r of v8f -> row r + 8*half, col l%16.
// ---------------------------------------------------------------------------
template <typename AT, typename OT>
__global__ void gemm_wmma_bf16(const AT* __restrict__ A,
                               const float* __restrict__ W,
                               const float* __restrict__ bias,
                               OT* __restrict__ C,
                               int M, int N, int K) {
  const int wave = blockIdx.x * (blockDim.x >> 5) + (threadIdx.x >> 5);
  const int lane = threadIdx.x & 31;
  const int tn   = N >> 4;
  const int mt   = wave / tn;
  const int nt   = wave % tn;
  if (mt * 16 >= M) return;

  const int m    = lane & 15;
  const int half = lane >> 4;

  v8f acc = {};
  const AT*    arow_base = A + (size_t)(mt * 16 + m) * K;
  const float* wrow_base = W + (size_t)(nt * 16 + m) * K + half * 16;

  for (int k0 = 0; k0 < K; k0 += 32) {
    v16bf a, b;
    const AT* arow = arow_base + k0;
#pragma unroll
    for (int j = 0; j < 8; ++j) {
      a[j]     = to_bf16(arow[half * 16 + j]);
      a[j + 8] = to_bf16(arow[half * 16 + 8 + j]);
    }
    const float* wrow = wrow_base + k0;
#pragma unroll
    for (int j = 0; j < 16; ++j) b[j] = (__bf16)wrow[j];
    acc = WMMA_BF16(a, b, acc);
  }

  const float bn = bias[nt * 16 + m];
#pragma unroll
  for (int r = 0; r < 8; ++r) {
    const int row = mt * 16 + r + 8 * half;
    float v = acc[r] + bn;
    if constexpr (__is_same(OT, float)) {
      C[(size_t)row * N + nt * 16 + m] = v;
    } else {
      C[(size_t)row * N + nt * 16 + m] = (__bf16)v;
    }
  }
}

// ---------------------------------------------------------------------------
// Pass 1: per (b, h, q-tile of 16) compute row logsumexp stats M, L over all N.
// One wave per tile; per-lane online softmax over its column subset, then a
// cross-lane LDS reduction.
// ---------------------------------------------------------------------------
__global__ void attn_ml_kernel(const __bf16* __restrict__ qp,
                               const __bf16* __restrict__ kp,
                               float* __restrict__ Mrow,
                               float* __restrict__ Lrow) {
  __shared__ float sm[4][32][8];
  __shared__ float sl[4][32][8];

  const int wslot = threadIdx.x >> 5;
  const int lane  = threadIdx.x & 31;
  const int wid   = blockIdx.x * 4 + wslot;

  const int qt = wid % (N_ / 16);
  const int h  = (wid / (N_ / 16)) % H_;
  const int b  = wid / ((N_ / 16) * H_);
  const int g  = h / HPG_;

  const int m = lane & 15, half = lane >> 4;

  const __bf16* qbase = qp + ((size_t)(b * N_ + qt * 16 + m)) * E_ + h * D_;
  v16bf qa0, qa1;
#pragma unroll
  for (int j = 0; j < 8; ++j) {
    qa0[j]     = qbase[half * 16 + j];
    qa0[j + 8] = qbase[half * 16 + 8 + j];
    qa1[j]     = qbase[32 + half * 16 + j];
    qa1[j + 8] = qbase[40 + half * 16 + j];
  }

  float pm[8], pl[8];
#pragma unroll
  for (int r = 0; r < 8; ++r) { pm[r] = -__builtin_inff(); pl[r] = 0.0f; }

  for (int nt = 0; nt < N_ / 16; ++nt) {
    const __bf16* kbase =
        kp + ((size_t)(b * N_ + nt * 16 + m)) * GD_ + g * D_ + half * 16;
    v16bf kb0, kb1;
#pragma unroll
    for (int j = 0; j < 16; ++j) { kb0[j] = kbase[j]; kb1[j] = kbase[32 + j]; }
    v8f s = {};
    s = WMMA_BF16(qa0, kb0, s);
    s = WMMA_BF16(qa1, kb1, s);
#pragma unroll
    for (int r = 0; r < 8; ++r) {
      float v = s[r] * SCALE_;
      if (v > pm[r]) { pl[r] *= __expf(pm[r] - v); pm[r] = v; }
      pl[r] += __expf(v - pm[r]);
    }
  }

#pragma unroll
  for (int r = 0; r < 8; ++r) { sm[wslot][lane][r] = pm[r]; sl[wslot][lane][r] = pl[r]; }
  __syncthreads();

  if (lane < 16) {
    const int halfr = lane >> 3, rr = lane & 7;
    float Mv = -__builtin_inff();
#pragma unroll
    for (int i = 0; i < 16; ++i)
      Mv = fmaxf(Mv, sm[wslot][halfr * 16 + i][rr]);
    float Lv = 0.0f;
#pragma unroll
    for (int i = 0; i < 16; ++i)
      Lv += sl[wslot][halfr * 16 + i][rr] * __expf(sm[wslot][halfr * 16 + i][rr] - Mv);
    const size_t idx = ((size_t)(b * H_ + h)) * N_ + qt * 16 + lane;
    Mrow[idx] = Mv;
    Lrow[idx] = Lv;
  }
}

// ---------------------------------------------------------------------------
// Pass 2a: head-summed attention matrix wsum[b,q,n]. One wave owns a 16x16
// (q,n) tile, loops the 4 KV groups (hoisting K fragments, shared by 4 heads
// under GQA) and the 4 heads within each. No atomics.
// ---------------------------------------------------------------------------
__global__ void attn_wsum_kernel(const __bf16* __restrict__ qp,
                                 const __bf16* __restrict__ kp,
                                 const float* __restrict__ Mrow,
                                 const float* __restrict__ Lrow,
                                 float* __restrict__ wsum) {
  const int wave = blockIdx.x * 4 + (threadIdx.x >> 5);
  const int lane = threadIdx.x & 31;

  const int nt = wave % (N_ / 16);
  const int qt = (wave / (N_ / 16)) % (N_ / 16);
  const int b  = wave / ((N_ / 16) * (N_ / 16));

  const int m = lane & 15, half = lane >> 4;

  float acc[8];
#pragma unroll
  for (int r = 0; r < 8; ++r) acc[r] = 0.0f;

  for (int g = 0; g < G_; ++g) {
    // K fragments shared by the 4 heads of this KV group.
    const __bf16* kbase =
        kp + ((size_t)(b * N_ + nt * 16 + m)) * GD_ + g * D_ + half * 16;
    v16bf kb0, kb1;
#pragma unroll
    for (int j = 0; j < 16; ++j) { kb0[j] = kbase[j]; kb1[j] = kbase[32 + j]; }

#pragma unroll
    for (int hh = 0; hh < HPG_; ++hh) {
      const int h = g * HPG_ + hh;
      const __bf16* qbase =
          qp + ((size_t)(b * N_ + qt * 16 + m)) * E_ + h * D_;
      v16bf qa0, qa1;
#pragma unroll
      for (int j = 0; j < 8; ++j) {
        qa0[j]     = qbase[half * 16 + j];
        qa0[j + 8] = qbase[half * 16 + 8 + j];
        qa1[j]     = qbase[32 + half * 16 + j];
        qa1[j + 8] = qbase[40 + half * 16 + j];
      }
      v8f s = {};
      s = WMMA_BF16(qa0, kb0, s);
      s = WMMA_BF16(qa1, kb1, s);

      const size_t mlbase = ((size_t)(b * H_ + h)) * N_ + qt * 16 + 8 * half;
#pragma unroll
      for (int r = 0; r < 8; ++r) {
        float Mv = Mrow[mlbase + r];
        float Lv = Lrow[mlbase + r];
        acc[r] += __expf(s[r] * SCALE_ - Mv) / Lv;
      }
    }
  }

#pragma unroll
  for (int r = 0; r < 8; ++r) {
    const size_t row = (size_t)b * N_ + qt * 16 + r + 8 * half;
    wsum[row * N_ + nt * 16 + m] = acc[r];
  }
}

// ---------------------------------------------------------------------------
// Pass 2b: O = softmax(QK^T) * V via WMMA. One wave per (b,h,q-tile).
// K/V tiles (32 keys x 64 d, bf16) are staged into wave-private LDS with
// GLOBAL_LOAD_ASYNC_TO_LDS_B128, double-buffered so the async copy of chunk
// nc+1 overlaps the WMMA of chunk nc. Probabilities are normalized with the
// precomputed M/L, transposed through LDS into an A-fragment, and accumulated
// against V B-fragments read from LDS.
// ---------------------------------------------------------------------------
__global__ void attn_o_kernel(const __bf16* __restrict__ qp,
                              const __bf16* __restrict__ kp,
                              const __bf16* __restrict__ vp,
                              const float* __restrict__ Mrow,
                              const float* __restrict__ Lrow,
                              __bf16* __restrict__ O) {
  __shared__ __bf16 kt[4][2][32 * 64];   // 32 KB: per-wave double-buffered K
  __shared__ __bf16 vt[4][2][32 * 64];   // 32 KB: per-wave double-buffered V
  __shared__ __bf16 wl[4][16 * 32];      //  4 KB: per-wave probability tile

  const int wslot = threadIdx.x >> 5;
  const int lane  = threadIdx.x & 31;
  const int wid   = blockIdx.x * 4 + wslot;

  const int qt = wid % (N_ / 16);
  const int h  = (wid / (N_ / 16)) % H_;
  const int b  = wid / ((N_ / 16) * H_);
  const int g  = h / HPG_;

  const int m = lane & 15, half = lane >> 4;

  // Per-lane coordinates for the async tile copy (32 rows x 128 bytes):
  // issue i covers rows i*4 + lane/8, 16 bytes per lane at (lane%8)*16.
  const int arow = lane >> 3;
  const int acol = (lane & 7) * 16;

  const __bf16* kgbase = kp + ((size_t)(b * N_)) * GD_ + g * D_;
  const __bf16* vgbase = vp + ((size_t)(b * N_)) * GD_ + g * D_;

  auto issue_stage = [&](int nc, int buf) {
    const int n0 = nc * 32;
    const unsigned kl = (unsigned)(unsigned long long)(&kt[wslot][buf][0]);
    const unsigned vl = (unsigned)(unsigned long long)(&vt[wslot][buf][0]);
#pragma unroll
    for (int i = 0; i < 8; ++i) {
      const int row = i * 4 + arow;
      const unsigned off = (unsigned)(row * 128 + acol);
      const char* ks = (const char*)(kgbase + (size_t)(n0 + row) * GD_) + acol;
      const char* vs = (const char*)(vgbase + (size_t)(n0 + row) * GD_) + acol;
      asm volatile("global_load_async_to_lds_b128 %0, %1, off"
                   :: "v"(kl + off), "v"(ks) : "memory");
      asm volatile("global_load_async_to_lds_b128 %0, %1, off"
                   :: "v"(vl + off), "v"(vs) : "memory");
    }
  };

  // Q fragments, loaded once.
  const __bf16* qbase = qp + ((size_t)(b * N_ + qt * 16 + m)) * E_ + h * D_;
  v16bf qa0, qa1;
#pragma unroll
  for (int j = 0; j < 8; ++j) {
    qa0[j]     = qbase[half * 16 + j];
    qa0[j + 8] = qbase[half * 16 + 8 + j];
    qa1[j]     = qbase[32 + half * 16 + j];
    qa1[j + 8] = qbase[40 + half * 16 + j];
  }

  float Mr[8], rL[8];
  const size_t mlbase = ((size_t)(b * H_ + h)) * N_ + qt * 16 + 8 * half;
#pragma unroll
  for (int r = 0; r < 8; ++r) {
    Mr[r] = Mrow[mlbase + r];
    rL[r] = 1.0f / Lrow[mlbase + r];
  }

  v8f accO[4];
#pragma unroll
  for (int dt = 0; dt < 4; ++dt) accO[dt] = (v8f){};

  constexpr int NC = N_ / 32;              // 64 chunks of 32 keys
  issue_stage(0, 0);

  for (int nc = 0; nc < NC; ++nc) {
    const int buf = nc & 1;
    if (nc + 1 < NC) {
      issue_stage(nc + 1, buf ^ 1);
      // 16 newer transfers outstanding; async loads complete in order, so
      // waiting <=16 guarantees stage nc has fully landed in LDS.
      asm volatile("s_wait_asynccnt 0x10" ::: "memory");
    } else {
      asm volatile("s_wait_asynccnt 0x0" ::: "memory");
    }

    const __bf16* ktile = &kt[wslot][buf][0];
    const __bf16* vtile = &vt[wslot][buf][0];

    // Two 16x16 score tiles -> one normalized 16x32 probability tile in LDS.
#pragma unroll
    for (int sub = 0; sub < 2; ++sub) {
      const __bf16* krow = ktile + (sub * 16 + m) * 64 + half * 16;
      v16bf kb0, kb1;
#pragma unroll
      for (int j = 0; j < 16; ++j) { kb0[j] = krow[j]; kb1[j] = krow[32 + j]; }
      v8f s = {};
      s = WMMA_BF16(qa0, kb0, s);
      s = WMMA_BF16(qa1, kb1, s);
#pragma unroll
      for (int r = 0; r < 8; ++r) {
        float w = __expf(s[r] * SCALE_ - Mr[r]) * rL[r];
        wl[wslot][(r + 8 * half) * 32 + sub * 16 + m] = (__bf16)w;
      }
    }
    // w tile is wave-private; DS ops are in-order within a wave, the barrier
    // only pins compiler scheduling between the stores above and loads below.
    __builtin_amdgcn_wave_barrier();

    // w as A-fragment (16x32 bf16) in contiguous k order (key = half*16 + j), same as vb.
    v16bf wa;
    const __bf16* wrow = &wl[wslot][m * 32];
#pragma unroll
    for (int j = 0; j < 8; ++j) {
      wa[j]     = wrow[half * 16 + j];
      wa[j + 8] = wrow[half * 16 + 8 + j];
    }

    // V B-fragments from LDS: 32 keys x 16 d-columns, 4 d-tiles cover D=64.
#pragma unroll
    for (int dt = 0; dt < 4; ++dt) {
      v16bf vb;
#pragma unroll
      for (int j = 0; j < 16; ++j)
        vb[j] = vtile[(half * 16 + j) * 64 + dt * 16 + m];
      accO[dt] = WMMA_BF16(wa, vb, accO[dt]);
    }
    __builtin_amdgcn_wave_barrier();
  }

  // O stored as [B,N,H*D] (feature = head*D + d), ready for the final GEMM.
#pragma unroll
  for (int dt = 0; dt < 4; ++dt) {
#pragma unroll
    for (int r = 0; r < 8; ++r) {
      const size_t row = (size_t)(b * N_ + qt * 16 + r + 8 * half);
      O[row * E_ + h * D_ + dt * 16 + m] = (__bf16)accO[dt][r];
    }
  }
}

// ---------------------------------------------------------------------------
extern "C" void kernel_launch(void* const* d_in, const int* in_sizes, int n_in,
                              void* d_out, int out_size, void* d_ws, size_t ws_size,
                              hipStream_t stream) {
  (void)in_sizes; (void)n_in; (void)out_size; (void)ws_size;

  const float* q   = (const float*)d_in[0];
  const float* k   = (const float*)d_in[1];
  const float* v   = (const float*)d_in[2];
  const float* Wq  = (const float*)d_in[3];
  const float* bq  = (const float*)d_in[4];
  const float* Wkv = (const float*)d_in[5];
  const float* bkv = (const float*)d_in[6];
  const float* Wo  = (const float*)d_in[7];
  const float* bo  = (const float*)d_in[8];

  float* out  = (float*)d_out;                 // [B,N,E]
  float* wsum = out + (size_t)BN_ * E_;        // [B,N,N]

  // Workspace carve-up (~21.5 MB).
  char* w = (char*)d_ws;
  __bf16* qpb = (__bf16*)w;  w += (size_t)BN_ * E_ * sizeof(__bf16);
  __bf16* kpb = (__bf16*)w;  w += (size_t)BN_ * GD_ * sizeof(__bf16);
  __bf16* vpb = (__bf16*)w;  w += (size_t)BN_ * GD_ * sizeof(__bf16);
  __bf16* Ob  = (__bf16*)w;  w += (size_t)BN_ * E_ * sizeof(__bf16);
  float*  Mrow = (float*)w;  w += (size_t)B_ * H_ * N_ * sizeof(float);
  float*  Lrow = (float*)w;

  const int THREADS = 128;                     // 4 waves / block
  auto blocks_for = [](int tiles) { return tiles / 4; };

  // 1) Projections (fp32 in -> bf16 out, WMMA bf16, fp32 accumulate).
  {
    int tiles = (BN_ / 16) * (E_ / 16);        // 16384
    gemm_wmma_bf16<float, __bf16><<<blocks_for(tiles), THREADS, 0, stream>>>(
        q, Wq, bq, qpb, BN_, E_, E_);
  }
  {
    int tiles = (BN_ / 16) * (GD_ / 16);       // 4096
    gemm_wmma_bf16<float, __bf16><<<blocks_for(tiles), THREADS, 0, stream>>>(
        k, Wkv, bkv, kpb, BN_, GD_, E_);
    gemm_wmma_bf16<float, __bf16><<<blocks_for(tiles), THREADS, 0, stream>>>(
        v, Wkv, bkv, vpb, BN_, GD_, E_);
  }

  // 2) Row softmax stats (logsumexp) per (b,h,q).
  {
    int waves = B_ * H_ * (N_ / 16);           // 4096
    attn_ml_kernel<<<blocks_for(waves), THREADS, 0, stream>>>(qpb, kpb, Mrow, Lrow);
  }

  // 3) Head-summed attention matrix.
  {
    int waves = B_ * (N_ / 16) * (N_ / 16);    // 32768
    attn_wsum_kernel<<<blocks_for(waves), THREADS, 0, stream>>>(
        qpb, kpb, Mrow, Lrow, wsum);
  }

  // 4) O = softmax(QK^T) V with async LDS staging, stored [B,N,E] bf16.
  {
    int waves = B_ * H_ * (N_ / 16);           // 4096
    attn_o_kernel<<<blocks_for(waves), THREADS, 0, stream>>>(
        qpb, kpb, vpb, Mrow, Lrow, Ob);
  }

  // 5) Output projection (bf16 A, fp32 W/out).
  {
    int tiles = (BN_ / 16) * (E_ / 16);        // 16384
    gemm_wmma_bf16<__bf16, float><<<blocks_for(tiles), THREADS, 0, stream>>>(
        Ob, Wo, bo, out, BN_, E_, E_);
  }
}
